// MultiHeadAttention_6966436954484
// MI455X (gfx1250) — matrix operand order repaired and checked by reading, not compiled, not run on hardware
//
#include <hip/hip_runtime.h>
#include <math.h>

// ---------------------------------------------------------------------------
// Fused MHA block for MI455X (gfx1250, wave32, WMMA bf16 16x16x32, f32 acc).
// B=4, S=2048, C=256 (head dim = 256), H=8.  ~172 GFLOP, compute bound.
// Pipeline: cvt(bf16) -> QKV proj -> flash attention -> FC+residual -> LN.
// ---------------------------------------------------------------------------

#define DEVINL __device__ __forceinline__

typedef __attribute__((ext_vector_type(16))) __bf16       v16bf;
typedef __attribute__((ext_vector_type(8)))  float        v8f;
typedef __attribute__((ext_vector_type(8)))  unsigned int v8u;

constexpr int Bn = 4, Sn = 2048, Cn = 256, Hn = 8, HCn = Hn * Cn;
constexpr float LN_EPS = 1e-5f;

DEVINL unsigned short f2bf(float f) {           // fp32 -> bf16, round-nearest-even
  unsigned int u = __float_as_uint(f);
  u += 0x7fffu + ((u >> 16) & 1u);
  return (unsigned short)(u >> 16);
}

// A fragment: 16(M) x 32(K) bf16 from row-major src (ld elems/row).
// Layout: lane<16 holds row M=lane, K=0..15;
// lane>=16 holds row M=lane-16, K=16..31. Pairs packed per dword.
DEVINL v16bf load_frag_a(const unsigned short* src, int ld, int row0, int k0, int lane) {
  const int ll = lane & 15, lh = lane >> 4;
  const unsigned short* p = src + (size_t)(row0 + ll) * ld + k0 + lh * 16;
  v8u u;
#pragma unroll
  for (int j = 0; j < 4; ++j) {
    u[j]     = *(const unsigned int*)(p + 2 * j);        // K = base + {0,2,4,6}
    u[j + 4] = *(const unsigned int*)(p + 8 + 2 * j);    // K = base + 8 + {0,2,4,6}
  }
  return __builtin_bit_cast(v16bf, u);
}

// B fragment: 32(K) x 16(N) bf16.  Memory holds column n contiguous in K
// ("transposed" layout: srcT[n*ld + k]).  Lane<16 -> K=0..15, lane>=16 -> K=16..31.
DEVINL v16bf load_frag_b(const unsigned short* srcT, int ld, int n0, int k0, int lane) {
  const int ll = lane & 15, lh = lane >> 4;
  const unsigned short* p = srcT + (size_t)(n0 + ll) * ld + k0 + lh * 16;
  v8u u;
#pragma unroll
  for (int j = 0; j < 8; ++j) u[j] = *(const unsigned int*)(p + 2 * j);
  return __builtin_bit_cast(v16bf, u);
}

DEVINL v8f wmma_bf16(v16bf a, v16bf b, v8f c) {
  // D(f32 16x16) = A(16x32 bf16) x B(32x16 bf16) + C
  return __builtin_amdgcn_wmma_f32_16x16x32_bf16(false, a, false, b, (short)0, c, false, false);
}

// --------------------------- conversion kernels ----------------------------

__global__ void cvt_kernel(const float* __restrict__ src, unsigned short* __restrict__ dst, int n) {
  int i = blockIdx.x * blockDim.x + threadIdx.x;
  if (i < n) dst[i] = f2bf(src[i]);
}

// dst[b][c][r] = src[b][r][c]   (transpose + bf16 convert; batch in blockIdx.y)
__global__ void cvt_t_kernel(const float* __restrict__ src, unsigned short* __restrict__ dst,
                             int rows, int cols) {
  int i = blockIdx.x * blockDim.x + threadIdx.x;
  if (i >= rows * cols) return;
  int r = i / cols, c = i % cols;
  size_t base = (size_t)blockIdx.y * rows * cols;
  dst[base + (size_t)c * rows + r] = f2bf(src[base + i]);
}

// ------------------------------ QKV projection -----------------------------
// One wave computes a 16x64 output tile of q, k and v for one (b,h),
// sharing the A (x) fragment.  q pre-scaled by 1/sqrt(C); v stored transposed.
__global__ void qkv_kernel(const unsigned short* __restrict__ xb,
                           const unsigned short* __restrict__ wqT,
                           const unsigned short* __restrict__ wkT,
                           const unsigned short* __restrict__ wvT,
                           const float* __restrict__ bq, const float* __restrict__ bk,
                           const float* __restrict__ bv,
                           unsigned short* __restrict__ qb,
                           unsigned short* __restrict__ kb,
                           unsigned short* __restrict__ vTb) {
  const int lane = threadIdx.x;
  const int h = blockIdx.y, b = blockIdx.z;
  const int mt = blockIdx.x >> 2;                  // S/16 row tiles
  const int nt4 = blockIdx.x & 3;                  // C/64 col groups
  const int s0 = mt * 16, n0 = nt4 * 64;

  const unsigned short* A = xb + (size_t)b * Sn * Cn;
  const unsigned short* WT[3] = { wqT + (size_t)h * Cn * Cn,
                                  wkT + (size_t)h * Cn * Cn,
                                  wvT + (size_t)h * Cn * Cn };
  v8f acc[3][4];
#pragma unroll
  for (int m = 0; m < 3; ++m)
#pragma unroll
    for (int nt = 0; nt < 4; ++nt) acc[m][nt] = (v8f)0.0f;

  for (int k0 = 0; k0 < Cn; k0 += 32) {
    v16bf a = load_frag_a(A, Cn, s0, k0, lane);
#pragma unroll
    for (int m = 0; m < 3; ++m)
#pragma unroll
      for (int nt = 0; nt < 4; ++nt) {
        v16bf bf = load_frag_b(WT[m], Cn, n0 + nt * 16, k0, lane);
        acc[m][nt] = wmma_bf16(a, bf, acc[m][nt]);
      }
  }

  const int ll = lane & 15, lh = lane >> 4;
  const float qscale = 0.0625f;                    // 1/sqrt(256)
#pragma unroll
  for (int nt = 0; nt < 4; ++nt)
#pragma unroll
    for (int r = 0; r < 8; ++r) {
      int row = s0 + lh * 8 + r;
      int col = n0 + nt * 16 + ll;
      size_t off = ((size_t)(b * Hn + h) * Sn + row) * Cn + col;
      qb[off] = f2bf((acc[0][nt][r] + bq[h * Cn + col]) * qscale);
      kb[off] = f2bf(acc[1][nt][r] + bk[h * Cn + col]);
      vTb[((size_t)(b * Hn + h) * Cn + col) * Sn + row] = f2bf(acc[2][nt][r] + bv[h * Cn + col]);
    }
}

// ------------------------- flash attention (1 wave) ------------------------
// One wave owns 16 query rows of one (b,h); iterates keys in 32-wide chunks.
// Per chunk: QK^T (16 wmma) -> online softmax -> P via LDS -> PV (16 wmma).
__global__ void attn_kernel(const unsigned short* __restrict__ qb,
                            const unsigned short* __restrict__ kb,
                            const unsigned short* __restrict__ vTb,
                            unsigned short* __restrict__ cat) {
  __shared__ unsigned short Pld[16 * 32];          // wave-private P tile (bf16)
  const int lane = threadIdx.x;
  const int h = blockIdx.y, b = blockIdx.z;
  const int s0 = blockIdx.x * 16;
  const int ll = lane & 15, lh = lane >> 4;

  const unsigned short* Q  = qb  + (size_t)(b * Hn + h) * Sn * Cn;
  const unsigned short* K  = kb  + (size_t)(b * Hn + h) * Sn * Cn;
  const unsigned short* VT = vTb + (size_t)(b * Hn + h) * Cn * Sn;

  v8f O[16];                                       // 16x256 f32 accumulator
#pragma unroll
  for (int d = 0; d < 16; ++d) O[d] = (v8f)0.0f;
  float Mrow[8], Lrow[8];
#pragma unroll
  for (int r = 0; r < 8; ++r) { Mrow[r] = -1e30f; Lrow[r] = 0.0f; }

#pragma unroll 1
  for (int t0 = 0; t0 < Sn; t0 += 32) {
    // --- scores: D0 = Q[16x256] @ K[t0..t0+15]^T, D1 = next 16 cols ---
    v8f D0 = (v8f)0.0f, D1 = (v8f)0.0f;
#pragma unroll
    for (int k0 = 0; k0 < Cn; k0 += 32) {
      v16bf a  = load_frag_a(Q, Cn, s0, k0, lane);
      v16bf b0 = load_frag_b(K, Cn, t0,      k0, lane);
      v16bf b1 = load_frag_b(K, Cn, t0 + 16, k0, lane);
      D0 = wmma_bf16(a, b0, D0);
      D1 = wmma_bf16(a, b1, D1);
    }
    // --- online softmax per row r (lanes 0-15: row r, lanes 16-31: row r+8) ---
    float rscale[8];
#pragma unroll
    for (int r = 0; r < 8; ++r) {
      float mx = fmaxf(D0[r], D1[r]);
#pragma unroll
      for (int m = 8; m >= 1; m >>= 1) mx = fmaxf(mx, __shfl_xor(mx, m, 32));
      float newM = fmaxf(Mrow[r], mx);
      rscale[r]  = __expf(Mrow[r] - newM);
      float p0   = __expf(D0[r] - newM);
      float p1   = __expf(D1[r] - newM);
      float rs = p0 + p1;
#pragma unroll
      for (int m = 8; m >= 1; m >>= 1) rs += __shfl_xor(rs, m, 32);
      Lrow[r] = Lrow[r] * rscale[r] + rs;
      Mrow[r] = newM;
      int mrow = lh * 8 + r;
      Pld[mrow * 32 + ll]      = f2bf(p0);
      Pld[mrow * 32 + 16 + ll] = f2bf(p1);
    }
    // rescale accumulator
#pragma unroll
    for (int d = 0; d < 16; ++d)
#pragma unroll
      for (int r = 0; r < 8; ++r) O[d][r] *= rscale[r];

    __syncthreads();                               // drain DS stores (1-wave block)

    // --- PV: O[16x256] += P[16x32] @ V[t0..t0+31, 0..255] ---
    v16bf pa = load_frag_a(Pld, 32, 0, 0, lane);
#pragma unroll
    for (int dt = 0; dt < 16; ++dt) {
      v16bf bf = load_frag_b(VT, Sn, dt * 16, t0, lane);
      O[dt] = wmma_bf16(pa, bf, O[dt]);
    }
  }

  // normalize + write into concat layout cat[b][s][h*C + d] (bf16)
#pragma unroll
  for (int dt = 0; dt < 16; ++dt)
#pragma unroll
    for (int r = 0; r < 8; ++r) {
      float val = O[dt][r] / Lrow[r];
      int row = s0 + lh * 8 + r;
      int col = dt * 16 + ll;
      cat[((size_t)b * Sn + row) * HCn + h * Cn + col] = f2bf(val);
    }
}

// ------------------------ FC + bias + residual (f32) -----------------------
__global__ void fc_kernel(const unsigned short* __restrict__ cat,
                          const unsigned short* __restrict__ wfcT,
                          const float* __restrict__ bfc,
                          const float* __restrict__ x,
                          float* __restrict__ fcOut) {
  const int lane = threadIdx.x;
  const int row0 = blockIdx.x * 16;                // over B*S = 8192 rows
  const int n0   = blockIdx.y * 64;                // over C = 256 cols
  v8f acc[4];
#pragma unroll
  for (int nt = 0; nt < 4; ++nt) acc[nt] = (v8f)0.0f;

#pragma unroll 1
  for (int k0 = 0; k0 < HCn; k0 += 32) {
    v16bf a = load_frag_a(cat, HCn, row0, k0, lane);
#pragma unroll
    for (int nt = 0; nt < 4; ++nt) {
      v16bf bf = load_frag_b(wfcT, HCn, n0 + nt * 16, k0, lane);
      acc[nt] = wmma_bf16(a, bf, acc[nt]);
    }
  }
  const int ll = lane & 15, lh = lane >> 4;
#pragma unroll
  for (int nt = 0; nt < 4; ++nt)
#pragma unroll
    for (int r = 0; r < 8; ++r) {
      int row = row0 + lh * 8 + r;
      int col = n0 + nt * 16 + ll;
      fcOut[(size_t)row * Cn + col] = acc[nt][r] + bfc[col] + x[(size_t)row * Cn + col];
    }
}

// ------------------------------- LayerNorm ---------------------------------
__global__ void ln_kernel(const float* __restrict__ fcOut,
                          const float* __restrict__ gamma,
                          const float* __restrict__ beta,
                          float* __restrict__ out) {
  const int wid = threadIdx.x >> 5, lane = threadIdx.x & 31;
  const int row = blockIdx.x * 8 + wid;            // 8 waves/block, 1 row/wave
  const float* p = fcOut + (size_t)row * Cn;
  float v[8], s = 0.f, s2 = 0.f;
#pragma unroll
  for (int i = 0; i < 8; ++i) {
    v[i] = p[lane + i * 32];
    s += v[i]; s2 += v[i] * v[i];
  }
#pragma unroll
  for (int m = 16; m >= 1; m >>= 1) { s += __shfl_xor(s, m, 32); s2 += __shfl_xor(s2, m, 32); }
  float mu = s * (1.0f / Cn);
  float var = s2 * (1.0f / Cn) - mu * mu;
  float inv = rsqrtf(var + LN_EPS);
  float* o = out + (size_t)row * Cn;
#pragma unroll
  for (int i = 0; i < 8; ++i) {
    int c = lane + i * 32;
    o[c] = (v[i] - mu) * inv * gamma[c] + beta[c];
  }
}

// ------------------------------- launcher ----------------------------------
extern "C" void kernel_launch(void* const* d_in, const int* in_sizes, int n_in,
                              void* d_out, int out_size, void* d_ws, size_t ws_size,
                              hipStream_t stream) {
  const float* x    = (const float*)d_in[0];
  const float* Wq   = (const float*)d_in[1];
  const float* bq   = (const float*)d_in[2];
  const float* Wk   = (const float*)d_in[3];
  const float* bk   = (const float*)d_in[4];
  const float* Wv   = (const float*)d_in[5];
  const float* bv   = (const float*)d_in[6];
  const float* Wfc  = (const float*)d_in[7];
  const float* bfc  = (const float*)d_in[8];
  const float* gam  = (const float*)d_in[9];
  const float* bet  = (const float*)d_in[10];
  float* out = (float*)d_out;

  // workspace carve-up (~144 MiB total)
  size_t off = 0;
  auto take = [&](size_t bytes) {
    void* p = (char*)d_ws + off;
    off += (bytes + 255) & ~(size_t)255;
    return p;
  };
  unsigned short* xb   = (unsigned short*)take((size_t)Bn * Sn * Cn * 2);
  unsigned short* wqT  = (unsigned short*)take((size_t)Hn * Cn * Cn * 2);
  unsigned short* wkT  = (unsigned short*)take((size_t)Hn * Cn * Cn * 2);
  unsigned short* wvT  = (unsigned short*)take((size_t)Hn * Cn * Cn * 2);
  unsigned short* wfcT = (unsigned short*)take((size_t)HCn * Cn * 2);
  unsigned short* qb   = (unsigned short*)take((size_t)Bn * Hn * Sn * Cn * 2);
  unsigned short* kb   = (unsigned short*)take((size_t)Bn * Hn * Sn * Cn * 2);
  unsigned short* vTb  = (unsigned short*)take((size_t)Bn * Hn * Cn * Sn * 2);
  unsigned short* cat  = (unsigned short*)take((size_t)Bn * Sn * HCn * 2);
  float*          fcT  = (float*)take((size_t)Bn * Sn * Cn * 4);
  (void)ws_size; (void)in_sizes; (void)n_in; (void)out_size;

  // 1) convert x to bf16; transpose-convert weights
  {
    int n = Bn * Sn * Cn;
    cvt_kernel<<<(n + 255) / 256, 256, 0, stream>>>(x, xb, n);
  }
  {
    dim3 g((Cn * Cn + 255) / 256, Hn);
    cvt_t_kernel<<<g, 256, 0, stream>>>(Wq, wqT, Cn, Cn);
    cvt_t_kernel<<<g, 256, 0, stream>>>(Wk, wkT, Cn, Cn);
    cvt_t_kernel<<<g, 256, 0, stream>>>(Wv, wvT, Cn, Cn);
  }
  {
    dim3 g((HCn * Cn + 255) / 256, 1);
    cvt_t_kernel<<<g, 256, 0, stream>>>(Wfc, wfcT, HCn, Cn);
  }

  // 2) QKV projections (q pre-scaled by 1/sqrt(C); v stored transposed)
  {
    dim3 g((Sn / 16) * (Cn / 64), Hn, Bn);         // 512 x 8 x 4 waves
    qkv_kernel<<<g, 32, 0, stream>>>(xb, wqT, wkT, wvT, bq, bk, bv, qb, kb, vTb);
  }

  // 3) flash attention -> concat layout
  {
    dim3 g(Sn / 16, Hn, Bn);                       // 128 x 8 x 4 waves
    attn_kernel<<<g, 32, 0, stream>>>(qb, kb, vTb, cat);
  }

  // 4) FC + bias + residual
  {
    dim3 g((Bn * Sn) / 16, Cn / 64);               // 512 x 4 waves
    fc_kernel<<<g, 32, 0, stream>>>(cat, wfcT, bfc, x, fcT);
  }

  // 5) LayerNorm
  ln_kernel<<<(Bn * Sn) / 8, 256, 0, stream>>>(fcT, gam, bet, out);
}
